// InducingPointLayer_2774548873986
// MI455X (gfx1250) — matrix operand order repaired and checked by reading, not compiled, not run on hardware
//
#include <hip/hip_runtime.h>
#include <hip/hip_fp16.h>

// ---------------------------------------------------------------------------
// InducingPointLayer for MI455X (gfx1250):
//   wave32, v_wmma_f32_16x16x32_f16 GEMMs, async global->LDS DMA tile staging
//   (ASYNCcnt double buffering), ds_load_b128 fragment gathers.
//   Block tile 128x64, 8 waves, 32x32 register tile per wave (4 WMMA / BK=32).
// ---------------------------------------------------------------------------

typedef _Float16 f16;
typedef _Float16 v16h __attribute__((ext_vector_type(16)));
typedef _Float16 v8h  __attribute__((ext_vector_type(8)));
typedef float    v8f  __attribute__((ext_vector_type(8)));

#define BDIM   64
#define ADIM   256
#define NIN    512
#define NPTS   2048
#define NROWS  (BDIM * ADIM)   // 16384
#define GLD    (2 * NPTS)      // 4096 (augmented [Kind|I] leading dim)

// ---------------------------------------------------------------------------
// Pass 1: masked per-feature sum / sum-of-squares + mask count.
// ---------------------------------------------------------------------------
__global__ void k_stats(const float* __restrict__ x, const int* __restrict__ mask,
                        float* __restrict__ nsum, float* __restrict__ s,
                        float* __restrict__ q) {
  int rb = blockIdx.x * 64;
  int t  = threadIdx.x;
  int f0 = t, f1 = t + 256;
  float s0 = 0.f, s1 = 0.f, q0 = 0.f, q1 = 0.f, mc = 0.f;
  for (int r = 0; r < 64; ++r) {
    int row = rb + r;
    float m  = (float)mask[row];
    float v0 = x[(size_t)row * NIN + f0];
    float v1 = x[(size_t)row * NIN + f1];
    s0 += v0 * m; q0 += v0 * v0 * m;
    s1 += v1 * m; q1 += v1 * v1 * m;
    mc += m;
  }
  atomicAdd(&s[f0], s0); atomicAdd(&q[f0], q0);
  atomicAdd(&s[f1], s1); atomicAdd(&q[f1], q1);
  if (t == 0) atomicAdd(nsum, mc);
}

// mean/std/scale finalize: mean = s/n, var = (q - s^2/n)/(n-1) (ddof=1)
__global__ void k_finalize(const float* __restrict__ gamma,
                           const float* __restrict__ nsum,
                           const float* __restrict__ s, const float* __restrict__ q,
                           float* __restrict__ mean, float* __restrict__ stdv,
                           float* __restrict__ ascale, float* __restrict__ bscale) {
  int f = blockIdx.x * blockDim.x + threadIdx.x;
  if (f >= NIN) return;
  float n   = *nsum;
  float mu  = s[f] / n;
  float var = (q[f] - s[f] * mu) / (n - 1.0f);
  float sd  = sqrtf(var) + 1e-5f;
  float g2  = gamma[f] * gamma[f] + 0.001f;
  mean[f] = mu; stdv[f] = sd;
  ascale[f] = 1.0f / (sd * g2);   // z = (x - mean) * ascale * m
  bscale[f] = 1.0f / g2;          // zi = ip * bscale
}

// z rows (f16 for WMMA) + f32 row squared norms (computed pre-rounding)
__global__ void k_build_z(const float* __restrict__ x, const int* __restrict__ mask,
                          const float* __restrict__ mean, const float* __restrict__ ascale,
                          f16* __restrict__ zh, float* __restrict__ znorm) {
  __shared__ float red[256];
  int row = blockIdx.x, t = threadIdx.x;
  float m = (float)mask[row];
  float acc = 0.f;
  for (int f = t; f < NIN; f += 256) {
    float v = (x[(size_t)row * NIN + f] - mean[f]) * ascale[f] * m;
    zh[(size_t)row * NIN + f] = (f16)v;
    acc += v * v;
  }
  red[t] = acc; __syncthreads();
  for (int k = 128; k > 0; k >>= 1) { if (t < k) red[t] += red[t + k]; __syncthreads(); }
  if (t == 0) znorm[row] = red[0];
}

__global__ void k_build_zi(const float* __restrict__ ip, const float* __restrict__ bscale,
                           f16* __restrict__ zih, float* __restrict__ zinorm) {
  __shared__ float red[256];
  int row = blockIdx.x, t = threadIdx.x;
  float acc = 0.f;
  for (int f = t; f < NIN; f += 256) {
    float v = ip[(size_t)row * NIN + f] * bscale[f];
    zih[(size_t)row * NIN + f] = (f16)v;
    acc += v * v;
  }
  red[t] = acc; __syncthreads();
  for (int k = 128; k > 0; k >>= 1) { if (t < k) red[t] += red[t + k]; __syncthreads(); }
  if (t == 0) zinorm[row] = red[0];
}

// Kinv (symmetric) -> f16 [N,K] copy from right half of augmented G
__global__ void k_kinvh(const float* __restrict__ G, f16* __restrict__ kinvh) {
  size_t i = (size_t)blockIdx.x * 256 + threadIdx.x;   // i = n*2048 + k
  size_t n = i >> 11, k = i & 2047;
  kinvh[i] = (f16)G[n * GLD + NPTS + k];               // Kinv[n][k] (symmetry)
}

// inducing_points^T -> f16 [N=512, K=2048]
__global__ void k_iph(const float* __restrict__ ip, f16* __restrict__ iph) {
  size_t i = (size_t)blockIdx.x * 256 + threadIdx.x;   // i = n*2048 + k
  size_t n = i >> 11, k = i & 2047;
  iph[i] = (f16)ip[k * NIN + n];
}

// ---------------------------------------------------------------------------
// WMMA GEMM, all operands f16 with K contiguous: A[M,K], B[N,K].
// Tiles staged via async global->LDS DMA (ASYNCcnt), double buffered.
// EPI: 0 = exp(-|{-2a}+rn+cn|)          -> Kx   (f32 + f16 copy)
//      1 = same + 0.05 on diag          -> Kind+0.05I into augmented G
//      2 = plain                        -> t    (f32 + f16 copy)
//      3 = (a*std+mean)*mask            -> x_new
// Block: 256 threads = 8 waves (4x2 over M x N), tile 128x64, BK=32.
// Wave register tile 32x32: 4 accumulators, A/B fragment reuse (4 wmma : 8 ds).
// ---------------------------------------------------------------------------
template <int EPI>
__global__ void k_gemm(const f16* __restrict__ A, int lda,
                       const f16* __restrict__ Bm, int ldb,
                       float* __restrict__ C, f16* __restrict__ Ch, int ldc, int Kdim,
                       const float* __restrict__ rown, const float* __restrict__ coln,
                       const float* __restrict__ stdv, const float* __restrict__ mean,
                       const int* __restrict__ mask) {
  // row stride 40 f16 = 80B (16B aligned); 16B async chunks land aligned.
  __shared__ __align__(16) f16 As[2][128][40];
  __shared__ __align__(16) f16 Bs[2][64][40];

  int tid  = threadIdx.x;
  int wave = tid >> 5, lane = tid & 31;
  int wm = wave & 3, wn = wave >> 2;         // 4 waves along M, 2 along N
  int half = lane >> 4, lr = lane & 15;
  int rowBase = blockIdx.y * 128;
  int colBase = blockIdx.x * 64;

  // async chunk mapping: chunk c -> row c>>2, f16 col (c&3)*8 (16B chunks)
  int ar0 = tid >> 2, ac0 = (tid & 3) << 3;          // A chunks tid, tid+256
  int ar1 = ar0 + 64;
  int br  = ar0, bc = ac0;                           // B chunk tid

  v8f acc00 = {}, acc01 = {}, acc10 = {}, acc11 = {};

  auto issue = [&](int s, int k0) {
    unsigned off, lds;
    off = (unsigned)(((rowBase + ar0) * lda + k0 + ac0) * 2);
    lds = (unsigned)(uintptr_t)&As[s][ar0][ac0];
    asm volatile("global_load_async_to_lds_b128 %0, %1, %2"
                 :: "v"(lds), "v"(off), "s"(A) : "memory");
    off = (unsigned)(((rowBase + ar1) * lda + k0 + ac0) * 2);
    lds = (unsigned)(uintptr_t)&As[s][ar1][ac0];
    asm volatile("global_load_async_to_lds_b128 %0, %1, %2"
                 :: "v"(lds), "v"(off), "s"(A) : "memory");
    off = (unsigned)(((colBase + br) * ldb + k0 + bc) * 2);
    lds = (unsigned)(uintptr_t)&Bs[s][br][bc];
    asm volatile("global_load_async_to_lds_b128 %0, %1, %2"
                 :: "v"(lds), "v"(off), "s"(Bm) : "memory");
  };

  int stages = Kdim >> 5;
  issue(0, 0);
  for (int i = 0; i < stages; ++i) {
    int cur = i & 1;
    if (i + 1 < stages) {
      issue(cur ^ 1, (i + 1) << 5);
      // in-order completion: <=3 outstanding means current stage's 3 are done
      asm volatile("s_wait_asynccnt 0x3" ::: "memory");
    } else {
      asm volatile("s_wait_asynccnt 0x0" ::: "memory");
    }
    __syncthreads();

    // A fragments (rows wm*32 + r*16 + lr): elems 0..7 = K 8*half+0..7, +16.
    v16h afrag[2];
#pragma unroll
    for (int r = 0; r < 2; ++r) {
      int am = wm * 32 + r * 16 + lr;
      v8h lo = *(const v8h*)&As[cur][am][half * 8];
      v8h hi = *(const v8h*)&As[cur][am][16 + half * 8];
#pragma unroll
      for (int e = 0; e < 8; ++e) { afrag[r][e] = lo[e]; afrag[r][e + 8] = hi[e]; }
    }
    // B fragments (cols wn*32 + c*16 + lr): elems 0..7 = K 8*half+0..7, +16.
    v16h bfrag[2];
#pragma unroll
    for (int c = 0; c < 2; ++c) {
      int bn = wn * 32 + c * 16 + lr;
      v8h lo = *(const v8h*)&Bs[cur][bn][half * 8];
      v8h hi = *(const v8h*)&Bs[cur][bn][16 + half * 8];
#pragma unroll
      for (int e = 0; e < 8; ++e) { bfrag[c][e] = lo[e]; bfrag[c][e + 8] = hi[e]; }
    }

    acc00 = __builtin_amdgcn_wmma_f32_16x16x32_f16(false, afrag[0], false, bfrag[0],
                                                   (short)0, acc00, false, false);
    acc01 = __builtin_amdgcn_wmma_f32_16x16x32_f16(false, afrag[0], false, bfrag[1],
                                                   (short)0, acc01, false, false);
    acc10 = __builtin_amdgcn_wmma_f32_16x16x32_f16(false, afrag[1], false, bfrag[0],
                                                   (short)0, acc10, false, false);
    acc11 = __builtin_amdgcn_wmma_f32_16x16x32_f16(false, afrag[1], false, bfrag[1],
                                                   (short)0, acc11, false, false);
    __syncthreads();   // done reading this buffer -> next iter may overwrite it
  }

  // C layout: VGPR v -> M = 8*half + v (within each 16x16 tile), N = lr.
#pragma unroll
  for (int r = 0; r < 2; ++r) {
#pragma unroll
    for (int c = 0; c < 2; ++c) {
      const v8f& acc = (r == 0) ? (c == 0 ? acc00 : acc01)
                                : (c == 0 ? acc10 : acc11);
#pragma unroll
      for (int v = 0; v < 8; ++v) {
        int grow = rowBase + wm * 32 + r * 16 + half * 8 + v;
        int gcol = colBase + wn * 32 + c * 16 + lr;
        float a = acc[v];
        size_t i0 = (size_t)grow * ldc + gcol;
        if (EPI == 0 || EPI == 1) {
          float d  = -2.0f * a + rown[grow] + coln[gcol];
          float vv = __expf(-fabsf(d));
          if (EPI == 1 && grow == gcol) vv += 0.05f;
          C[i0] = vv;
          if (EPI == 0) Ch[i0] = (f16)vv;
        } else if (EPI == 2) {
          C[i0] = a;
          Ch[i0] = (f16)a;
        } else {
          float m = (float)mask[grow];
          C[i0] = (a * stdv[gcol] + mean[gcol]) * m;
        }
      }
    }
  }
}

// ---------------------------------------------------------------------------
// Gauss-Jordan on augmented G = [Kind+0.05I | I], 2048x4096. SPD -> no pivoting.
// ---------------------------------------------------------------------------
__global__ void k_eye(float* __restrict__ G) {
  int i = blockIdx.y;
  int j = blockIdx.x * 256 + threadIdx.x;            // 0..2047
  G[(size_t)i * GLD + NPTS + j] = (i == j) ? 1.0f : 0.0f;
}

__global__ void k_gj_prep(const float* __restrict__ G, float* __restrict__ colfac,
                          float* __restrict__ rowbuf, int k) {
  int t = blockIdx.x * 256 + threadIdx.x;
  if (t < NPTS) {
    colfac[t] = G[(size_t)t * GLD + k];
  } else {
    int j = t - NPTS;
    if (j < GLD) {
      float piv = G[(size_t)k * GLD + k];
      rowbuf[j] = G[(size_t)k * GLD + j] / piv;
    }
  }
}

__global__ void k_gj_update(float* __restrict__ G, const float* __restrict__ colfac,
                            const float* __restrict__ rowbuf, int k) {
  int i = blockIdx.y;
  int j = blockIdx.x * 256 + threadIdx.x;
  if (j < k) return;                                 // converged identity cols
  size_t idx = (size_t)i * GLD + j;
  if (i == k) G[idx] = rowbuf[j];
  else        G[idx] -= colfac[i] * rowbuf[j];
}

// var_out = (1 - rowsum(t .* Kx)) * m   (f32 operands: cancellation-sensitive)
__global__ void k_var(const float* __restrict__ t, const float* __restrict__ kx,
                      const int* __restrict__ mask, float* __restrict__ vout) {
  __shared__ float red[256];
  int row = blockIdx.x, tt = threadIdx.x;
  float acc = 0.f;
  for (int j = tt; j < NPTS; j += 256)
    acc += t[(size_t)row * NPTS + j] * kx[(size_t)row * NPTS + j];
  red[tt] = acc; __syncthreads();
  for (int k = 128; k > 0; k >>= 1) { if (tt < k) red[tt] += red[tt + k]; __syncthreads(); }
  if (tt == 0) vout[row] = (1.0f - red[0]) * (float)mask[row];
}

// ---------------------------------------------------------------------------
extern "C" void kernel_launch(void* const* d_in, const int* in_sizes, int n_in,
                              void* d_out, int out_size, void* d_ws, size_t ws_size,
                              hipStream_t stream) {
  const float* x     = (const float*)d_in[0];
  const int*   mask  = (const int*)d_in[1];
  const float* ip    = (const float*)d_in[2];
  const float* gamma = (const float*)d_in[3];

  float* out_x = (float*)d_out;                 // [16384, 512]
  float* out_v = out_x + (size_t)NROWS * NIN;   // [16384]

  // workspace layout (byte offsets, all 4KB aligned); total ~445 MB
  char* W = (char*)d_ws;
  float* w_n    = (float*)(W + 0);
  float* w_s    = (float*)(W + 4096);
  float* w_q    = (float*)(W + 8192);
  float* w_mean = (float*)(W + 12288);
  float* w_std  = (float*)(W + 16384);
  float* w_a    = (float*)(W + 20480);
  float* w_b    = (float*)(W + 24576);
  float* w_zn   = (float*)(W + 28672);          // 16384 f32
  float* w_zin  = (float*)(W + 94208);          // 2048 f32
  float* w_cf   = (float*)(W + 102400);         // 2048 f32
  float* w_rb   = (float*)(W + 110592);         // 4096 f32
  f16*   w_zh   = (f16*)  (W + 131072);         // 16384x512 f16
  f16*   w_zih  = (f16*)  (W + 16908288ull);    // 2048x512 f16
  float* w_kx   = (float*)(W + 18989056ull);    // 16384x2048 f32
  f16*   w_kxh  = (f16*)  (W + 153206784ull);   // 16384x2048 f16
  float* w_G    = (float*)(W + 220315648ull);   // 2048x4096 f32
  f16*   w_kih  = (f16*)  (W + 253870080ull);   // 2048x2048 f16
  float* w_t    = (float*)(W + 262258688ull);   // 16384x2048 f32
  f16*   w_th   = (f16*)  (W + 396476416ull);   // 16384x2048 f16
  f16*   w_iph  = (f16*)  (W + 463585280ull);   // 512x2048 f16

  // zero the atomic accumulators (nsum, s, q)
  hipMemsetAsync(d_ws, 0, 12288, stream);

  // stats -> mean/std/scales
  k_stats<<<NROWS / 64, 256, 0, stream>>>(x, mask, w_n, w_s, w_q);
  k_finalize<<<2, 256, 0, stream>>>(gamma, w_n, w_s, w_q, w_mean, w_std, w_a, w_b);

  // z, zi (f16) + norms; ip^T (f16)
  k_build_z<<<NROWS, 256, 0, stream>>>(x, mask, w_mean, w_a, w_zh, w_zn);
  k_build_zi<<<NPTS, 256, 0, stream>>>(ip, w_b, w_zih, w_zin);
  k_iph<<<(NIN * NPTS) / 256, 256, 0, stream>>>(ip, w_iph);

  // Kx = exp(-|dist(z, zi)|)           [16384 x 2048], K = 512
  k_gemm<0><<<dim3(NPTS / 64, NROWS / 128), 256, 0, stream>>>(
      w_zh, NIN, w_zih, NIN, w_kx, w_kxh, NPTS, NIN,
      w_zn, w_zin, nullptr, nullptr, nullptr);

  // Kind + 0.05I into left half of augmented G; right half = I
  k_gemm<1><<<dim3(NPTS / 64, NPTS / 128), 256, 0, stream>>>(
      w_zih, NIN, w_zih, NIN, w_G, nullptr, GLD, NIN,
      w_zin, w_zin, nullptr, nullptr, nullptr);
  k_eye<<<dim3(NPTS / 256, NPTS), 256, 0, stream>>>(w_G);

  // Gauss-Jordan elimination (SPD, no pivoting)
  for (int k = 0; k < NPTS; ++k) {
    k_gj_prep<<<(NPTS + GLD) / 256, 256, 0, stream>>>(w_G, w_cf, w_rb, k);
    k_gj_update<<<dim3(GLD / 256, NPTS), 256, 0, stream>>>(w_G, w_cf, w_rb, k);
  }
  k_kinvh<<<(NPTS * NPTS) / 256, 256, 0, stream>>>(w_G, w_kih);

  // t = Kx @ Kinv                      [16384 x 2048], K = 2048
  k_gemm<2><<<dim3(NPTS / 64, NROWS / 128), 256, 0, stream>>>(
      w_kxh, NPTS, w_kih, NPTS, w_t, w_th, NPTS, NPTS,
      nullptr, nullptr, nullptr, nullptr, nullptr);

  // x_new = ((t @ IP) * std + mean) * m  [16384 x 512], K = 2048
  k_gemm<3><<<dim3(NIN / 64, NROWS / 128), 256, 0, stream>>>(
      w_th, NPTS, w_iph, NPTS, out_x, nullptr, NIN, NPTS,
      nullptr, nullptr, w_std, w_mean, mask);

  // var_out = (1 - rowsum(t .* Kx)) * m
  k_var<<<NROWS, 256, 0, stream>>>(w_t, w_kx, mask, out_v);

  (void)in_sizes; (void)n_in; (void)out_size; (void)ws_size;
}
